// brutesolver_65524021068210
// MI455X (gfx1250) — 8-bit matrix operand order repaired and checked by reading, compiled, not run on hardware
//
#include <hip/hip_runtime.h>

typedef __attribute__((ext_vector_type(8))) int v8i;

#define FW 11
#define PAD 5

__global__ void bs_zero_ws(float* __restrict__ ws) {
  if (threadIdx.x < 16) ws[threadIdx.x] = 0.0f;
}

__device__ __forceinline__ int pack4(unsigned d0, unsigned d1,
                                     unsigned d2, unsigned d3) {
  unsigned lo = __builtin_amdgcn_perm(d1, d0, 0x00000703u);
  unsigned hi = __builtin_amdgcn_perm(d3, d2, 0x07030000u);
  return (int)(lo | hi);
}

__global__ __launch_bounds__(256) void bs_corr_wmma(
    const float* __restrict__ filters,
    const float* __restrict__ order,
    const float* __restrict__ wcorr,
    const int*   __restrict__ x,
    float*       __restrict__ ysum,
    int perms) {
  const int lane = threadIdx.x & 31;
  const int g    = lane >> 4;
  const int n    = lane & 15;
  const int waveId   = (blockIdx.x * blockDim.x + threadIdx.x) >> 5;
  const int numWaves = (gridDim.x * blockDim.x) >> 5;
  const int fullTiles = perms >> 4;
  const int tailCnt   = perms & 15;

  __shared__ float sy[16];
  if (threadIdx.x < 16) sy[threadIdx.x] = 0.0f;
  __syncthreads();

  unsigned occ = 0;
#pragma unroll
  for (int i = 0; i < 16; ++i) occ |= (x[i] == 1 ? 1u : 0u) << i;

  const int py = n >> 2, px = n & 3;
  v8i bmat;
#pragma unroll
  for (int j = 0; j < 8; ++j) {
    int kb = ((j >> 1) << 4) + ((j & 1) << 2) + (g << 3);
    unsigned wrd = 0;
#pragma unroll
    for (int t = 0; t < 4; ++t) {
      int k  = kb + t;
      int dy = k / FW, dx = k % FW;
      int yy = py + dy - PAD, xx = px + dx - PAD;
      unsigned bit = 0;
      if (yy >= 0 && yy < 4 && xx >= 0 && xx < 4)
        bit = (occ >> (yy * 4 + xx)) & 1u;
      wrd |= bit << (8 * t);
    }
    bmat[j] = (int)wrd;
  }

  float local = 0.0f;

  for (int tile = waveId; tile < fullTiles; tile += numWaves) {
    const int base = tile << 4;

    const float* fr = filters + (long long)(base + (lane & 15)) * 66;

    {
      long long nrow = (long long)base + (long long)numWaves * 16 + (lane & 15);
      if (nrow < perms) __builtin_prefetch(filters + nrow * 66, 0, 1);
    }

    v8i amat;
#pragma unroll
    for (int j = 0; j < 8; ++j) {
      int ka = ((j >> 1) << 4) + ((j & 1) << 2) + (g << 3);
      uint2 lo = *(const uint2*)(fr + ka);
      uint2 hi = *(const uint2*)(fr + ka + 2);
      amat[j] = pack4(lo.x, lo.y, hi.x, hi.y);
    }

    v8i cz = {0, 0, 0, 0, 0, 0, 0, 0};
    v8i c = __builtin_amdgcn_wmma_i32_16x16x64_iu8(
         false, amat,  false, bmat, cz,
         false,  false);

    const int mBase = base + (g << 3);
    float4 o0 = *(const float4*)(order + mBase);
    float4 o1 = *(const float4*)(order + mBase + 4);
    float4 w0 = *(const float4*)(wcorr + mBase);
    float4 w1 = *(const float4*)(wcorr + mBase + 4);
    float of[8] = {o0.x, o0.y, o0.z, o0.w, o1.x, o1.y, o1.z, o1.w};
    float wf[8] = {w0.x, w0.y, w0.z, w0.w, w1.x, w1.y, w1.z, w1.w};
#pragma unroll
    for (int r = 0; r < 8; ++r)
      local += (c[r] == 63 * (int)of[r]) ? wf[r] : 0.0f;
  }

  if (tailCnt && waveId == 0) {
    const int base = fullTiles << 4;
    int row = base + (lane & 15);
    row = row < perms ? row : perms - 1;
    const float* fr = filters + (long long)row * 66;

    v8i amat;
#pragma unroll
    for (int j = 0; j < 8; ++j) {
      int ka = ((j >> 1) << 4) + ((j & 1) << 2) + (g << 3);
      uint2 lo = *(const uint2*)(fr + ka);
      uint2 hi = *(const uint2*)(fr + ka + 2);
      amat[j] = pack4(lo.x, lo.y, hi.x, hi.y);
    }

    v8i cz = {0, 0, 0, 0, 0, 0, 0, 0};
    v8i c = __builtin_amdgcn_wmma_i32_16x16x64_iu8(
        false, amat, false, bmat, cz, false, false);

    const int mBase = base + (g << 3);
#pragma unroll
    for (int r = 0; r < 8; ++r) {
      int idx = mBase + r;
      if (idx < perms && c[r] == 63 * (int)order[idx]) local += wcorr[idx];
    }
  }

  atomicAdd(&sy[n], local);
  __syncthreads();
  if (threadIdx.x < 16) atomicAdd(&ysum[threadIdx.x], sy[threadIdx.x]);
}

__global__ void bs_finalize(const float* __restrict__ ysum,
                            const float* __restrict__ den_b,
                            float* __restrict__ out) {
  int n = threadIdx.x;
  if (n < 16) {
    float yv = ysum[n] + den_b[0];
    out[n]       = -100.0f;
    out[16 + n]  = -yv;
    out[32 + n]  = yv;
  }
}

extern "C" void kernel_launch(void* const* d_in, const int* in_sizes, int n_in,
                              void* d_out, int out_size, void* d_ws, size_t ws_size,
                              hipStream_t stream) {
  const int*   x       = (const int*)d_in[0];
  const float* filters = (const float*)d_in[1];
  const float* order   = (const float*)d_in[2];
  const float* wcorr   = (const float*)d_in[3];
  const float* den_b   = (const float*)d_in[5];
  const int perms = in_sizes[2];

  float* ysum = (float*)d_ws;

  bs_zero_ws<<<1, 32, 0, stream>>>(ysum);
  bs_corr_wmma<<<1024, 256, 0, stream>>>(filters, order, wcorr, x, ysum, perms);
  bs_finalize<<<1, 32, 0, stream>>>(ysum, den_b, (float*)d_out);
}
